// FA_GNN_89481348645568
// MI455X (gfx1250) — matrix operand order repaired and checked by reading, not compiled, not run on hardware
//
#include <hip/hip_runtime.h>
#include <hip/hip_bf16.h>

typedef __bf16 bf16_t;
typedef __attribute__((ext_vector_type(16))) __bf16 v16bf;
typedef __attribute__((ext_vector_type(8)))  float  v8f;
typedef __attribute__((ext_vector_type(4)))  int    v4i;
typedef __attribute__((ext_vector_type(8)))  int    v8i;

#define NN  5      // nodes per graph
#define NO  8      // frames
#define FD  96     // hidden features
#define NTN 6      // 96/16 N-tiles

__device__ __forceinline__ float silu_f(float x) {
    return x * (1.0f / (1.0f + __expf(-x)));
}
// OPS[o][j]: o=0 -> (+,+,+) ... bit2=j0, bit1=j1, bit0=j2
__device__ __forceinline__ float sgn_oj(int o, int j) {
    return ((o >> (2 - j)) & 1) ? -1.0f : 1.0f;
}

// ---------------- Tensor Data Mover: stage a 1-D block into LDS -------------
// Builds a 2-group D# (count=1, type=2 "image", data_size=4B, single row of
// `ndwords` elements) and issues TENSOR_LOAD_TO_LDS. Caller must be wave-
// uniform; completion via s_wait_tensorcnt.
__device__ __forceinline__ void tdm_load_1d(const void* gsrc, void* lds_dst,
                                            unsigned ndwords) {
    unsigned long long ga = (unsigned long long)(uintptr_t)gsrc;
    unsigned lds = (unsigned)(uintptr_t)lds_dst;   // flat LDS addr[31:0] = offset
    v4i g0;
    g0[0] = 1;                                     // count=1, user descriptor
    g0[1] = (int)lds;                              // lds_addr
    g0[2] = (int)(ga & 0xFFFFFFFFu);               // global_addr[31:0]
    g0[3] = (int)((ga >> 32) & 0x1FFFFFFu) | (int)0x80000000u; // addr[56:32]|type=2
    v8i g1;
    g1[0] = 0x00020000;                            // data_size=4B, no mask/pad
    g1[1] = (int)((ndwords & 0xFFFFu) << 16);      // tensor_dim0[15:0]
    g1[2] = (int)((ndwords >> 16) | (1u << 16));   // tensor_dim0[31:16], tdim1=1
    g1[3] = (int)((ndwords & 0xFFFFu) << 16);      // tile_dim0
    g1[4] = 0;                                     // tile_dim1/2 unused
    g1[5] = (int)ndwords;                          // tensor_dim0_stride[31:0]
    g1[6] = 0;
    g1[7] = 0;
    asm volatile("tensor_load_to_lds %0, %1" : : "s"(g0), "s"(g1) : "memory");
}

// ---------- 3x3 symmetric eigensolver (cyclic Jacobi), ascending eigenvalues,
// eigenvectors as COLUMNS of V (matches jnp.linalg.eigh convention). ----------
__device__ void eigh3_asc(float A[3][3], float V[3][3]) {
    V[0][0]=1.f; V[0][1]=0.f; V[0][2]=0.f;
    V[1][0]=0.f; V[1][1]=1.f; V[1][2]=0.f;
    V[2][0]=0.f; V[2][1]=0.f; V[2][2]=1.f;
#define JROT(p,q,r) do {                                               \
    float apq = A[p][q];                                               \
    if (fabsf(apq) > 1e-30f) {                                         \
        float th = (A[q][q] - A[p][p]) / (2.0f * apq);                 \
        float t  = copysignf(1.0f, th) / (fabsf(th) + sqrtf(th*th + 1.0f)); \
        float c  = 1.0f / sqrtf(t*t + 1.0f);                           \
        float s  = t * c;                                              \
        float app = A[p][p] - t*apq;                                   \
        float aqq = A[q][q] + t*apq;                                   \
        float arp = c*A[r][p] - s*A[r][q];                             \
        float arq = s*A[r][p] + c*A[r][q];                             \
        A[p][p]=app; A[q][q]=aqq; A[p][q]=0.f; A[q][p]=0.f;            \
        A[r][p]=arp; A[p][r]=arp; A[r][q]=arq; A[q][r]=arq;            \
        _Pragma("unroll")                                              \
        for (int k=0;k<3;++k) {                                        \
            float vp = c*V[k][p] - s*V[k][q];                          \
            float vq = s*V[k][p] + c*V[k][q];                          \
            V[k][p]=vp; V[k][q]=vq;                                    \
        }                                                              \
    } } while(0)
    for (int sw = 0; sw < 6; ++sw) { JROT(0,1,2); JROT(0,2,1); JROT(1,2,0); }
#undef JROT
#define CSWAP(i,j) do { if (A[i][i] > A[j][j]) {                       \
    float tmp = A[i][i]; A[i][i]=A[j][j]; A[j][j]=tmp;                 \
    _Pragma("unroll")                                                  \
    for (int k=0;k<3;++k){ float tv=V[k][i]; V[k][i]=V[k][j]; V[k][j]=tv; } \
    } } while(0)
    CSWAP(0,1); CSWAP(1,2); CSWAP(0,1);
#undef CSWAP
}

// ---------------- frame construction from the 5 input points ----------------
__global__ void k_make_frame(const float* __restrict__ hin,
                             float* __restrict__ Fb, float* __restrict__ ctr, int B) {
    int b = blockIdx.x * blockDim.x + threadIdx.x;
    if (b >= B) return;
    float s1[3] = {0.f,0.f,0.f};
    float s2[6] = {0.f,0.f,0.f,0.f,0.f,0.f};   // xx xy xz yy yz zz
    for (int n = 0; n < NN; ++n) {
        const float* p = hin + (size_t)(b*NN + n) * 6;
        float x = p[0], y = p[1], z = p[2];
        s1[0]+=x; s1[1]+=y; s1[2]+=z;
        s2[0]+=x*x; s2[1]+=x*y; s2[2]+=x*z; s2[3]+=y*y; s2[4]+=y*z; s2[5]+=z*z;
    }
    float c0 = s1[0]/NN, c1 = s1[1]/NN, c2 = s1[2]/NN;
    float A[3][3], V[3][3];
    A[0][0]=s2[0]-NN*c0*c0; A[0][1]=s2[1]-NN*c0*c1; A[0][2]=s2[2]-NN*c0*c2;
    A[1][0]=A[0][1];        A[1][1]=s2[3]-NN*c1*c1; A[1][2]=s2[4]-NN*c1*c2;
    A[2][0]=A[0][2];        A[2][1]=A[1][2];        A[2][2]=s2[5]-NN*c2*c2;
    eigh3_asc(A, V);
    float* fb = Fb + (size_t)b*9;
    #pragma unroll
    for (int i=0;i<3;++i)
        #pragma unroll
        for (int j=0;j<3;++j) fb[i*3+j] = V[i][j];
    ctr[b*3+0]=c0; ctr[b*3+1]=c1; ctr[b*3+2]=c2;
}

// -------- apply 8 frames to (pos,vel) and embed 6->96 in one pass -----------
__global__ void k_frame_embed(const float* __restrict__ hin,
                              const float* __restrict__ Fb, const float* __restrict__ ctr,
                              const float* __restrict__ embw, const float* __restrict__ embb,
                              float* __restrict__ h, int B, int ROWS) {
    int tid = blockIdx.x * blockDim.x + threadIdx.x;
    if (tid >= ROWS) return;
    int o   = tid / (B * NN);
    int rem = tid - o * (B * NN);
    int b   = rem / NN;
    int n   = rem - b * NN;
    const float* node = hin + (size_t)(b*NN + n) * 6;
    float xc[3], vv[3];
    #pragma unroll
    for (int j=0;j<3;++j) { xc[j] = node[j] - ctr[b*3+j]; vv[j] = node[3+j]; }
    const float* V = Fb + (size_t)b*9;
    float feat[6];
    #pragma unroll
    for (int i=0;i<3;++i) {
        float s = sgn_oj(o, i);
        feat[i]   = s * (V[0*3+i]*xc[0] + V[1*3+i]*xc[1] + V[2*3+i]*xc[2]);
        feat[3+i] = s * (V[0*3+i]*vv[0] + V[1*3+i]*vv[1] + V[2*3+i]*vv[2]);
    }
    float* out = h + (size_t)tid * FD;
    for (int c = 0; c < FD; ++c) {
        float acc = embb[c];
        #pragma unroll
        for (int k=0;k<6;++k) acc += feat[k] * embw[k*FD + c];
        out[c] = acc;
    }
}

__global__ void k_zero(float* __restrict__ p, int n) {
    int i = blockIdx.x * blockDim.x + threadIdx.x;
    if (i < n) p[i] = 0.f;
}

// ---- pack W[K,96] f32 -> bf16 in the exact B-matrix register layout --------
// B 32x16 layout: lanes 0-15 hold K={0..7,16..23}, lanes 16-31 K={8..15,24..31}
__global__ void k_pack_w(const float* __restrict__ src, bf16_t* __restrict__ dst,
                         int K, int KT) {
    int idx = blockIdx.x * blockDim.x + threadIdx.x;
    int total = KT * NTN * 512;
    if (idx >= total) return;
    int i    = idx & 15;
    int lane = (idx >> 4) & 31;
    int t    = idx >> 9;
    int nt   = t % NTN;
    int kt   = t / NTN;
    int k    = kt*32 + (lane >> 4)*8 + (i & 7) + ((i >> 3) << 4);
    int n    = nt*16 + (lane & 15);
    dst[idx] = (k < K) ? (bf16_t)src[(size_t)k*FD + n] : (bf16_t)0.0f;
}

// A 16x32 bf16 layout: lane<16: M=lane, K = {0..7, 16..23}; lane>=16: M=lane-16,
// K = {8..15, 24..31}. Given src -> feature (kt*32 + hi*8), fill 16 elems.
__device__ __forceinline__ void load_a_tile(const float* __restrict__ src, v16bf& a) {
    #pragma unroll
    for (int j=0;j<8;++j) { a[j] = (bf16_t)src[j]; a[8+j] = (bf16_t)src[16+j]; }
}
__device__ __forceinline__ v16bf lds_b_tile(const bf16_t* sW, int kt, int nt, int lane) {
    return *(const v16bf*)(sW + (((size_t)kt*NTN + nt)*32 + lane)*16);
}

// --------------------------- fused edge GCL ---------------------------------
// Persistent waves: weights staged once per block into LDS via TDM; each wave
// processes `trips` 16-edge tiles.  m1 = silu([h[row]|h[col]|ea]@W1+b1);
// m2 = silu(m1@W2+b2); atomicAdd into agg[row].
__global__ void __launch_bounds__(128)
k_edge_gcl(const float* __restrict__ h,
           const int* __restrict__ erow, const int* __restrict__ ecol,
           const float* __restrict__ ea,
           const bf16_t* __restrict__ W1, const float* __restrict__ b1,
           const bf16_t* __restrict__ W2, const float* __restrict__ b2,
           float* __restrict__ agg, int E, int NH,
           int ntiles, int nwaves, int trips) {
    __shared__ bf16_t sW1[7*NTN*512];   // 43008 B
    __shared__ bf16_t sW2[3*NTN*512];   // 18432 B
    __shared__ float  sm[4][16][FD];    // 24576 B
    __shared__ int    srow[4][16];
    __shared__ int    scol[4][16];
    if (threadIdx.x < 32) {             // wave 0: TDM-stage both weight blocks
        tdm_load_1d(W1, sW1, (7*NTN*512*2) / 4);
        tdm_load_1d(W2, sW2, (3*NTN*512*2) / 4);
        __builtin_amdgcn_s_wait_tensorcnt(0);
    }
    __syncthreads();
    int w    = threadIdx.x >> 5;
    int lane = threadIdx.x & 31;
    int gw   = blockIdx.x * 4 + w;
    int m    = lane & 15;
    int hi   = lane >> 4;
    for (int t = 0; t < trips; ++t) {
        int tile = gw + t * nwaves;
        bool active = tile < ntiles;
        if (active && lane < 16) {
            int e8 = tile*16 + lane;
            int o  = e8 / E;
            int e0 = e8 - o*E;
            srow[w][lane] = erow[e0] + o*NH;
            scol[w][lane] = ecol[e0] + o*NH;
        }
        __syncthreads();
        if (active) {
            int ro = srow[w][m];
            int co = scol[w][m];
            int koff = hi * 8;
            v16bf A[7];
            #pragma unroll
            for (int kt=0; kt<3; ++kt) {
                load_a_tile(h + (size_t)ro*FD + kt*32 + koff, A[kt]);
                load_a_tile(h + (size_t)co*FD + kt*32 + koff, A[3+kt]);
            }
            #pragma unroll
            for (int j=0;j<16;++j) A[6][j] = (bf16_t)0.0f;
            if (hi == 0) {                       // edge_attr at padded K=192,193
                int e8 = tile*16 + m;
                int o  = e8 / E;
                int e0 = e8 - o*E;
                A[6][0] = (bf16_t)ea[(size_t)e0*2 + 0];
                A[6][1] = (bf16_t)ea[(size_t)e0*2 + 1];
            }
            #pragma unroll
            for (int nt=0; nt<NTN; ++nt) {
                v8f c = {};
                #pragma unroll
                for (int kt=0; kt<7; ++kt)
                    c = __builtin_amdgcn_wmma_f32_16x16x32_bf16(
                            false, A[kt], false, lds_b_tile(sW1, kt, nt, lane),
                            (short)0, c, false, false);
                int n = nt*16 + m;
                float bb = b1[n];
                #pragma unroll
                for (int r=0;r<8;++r) sm[w][r + hi*8][n] = silu_f(c[r] + bb);
            }
        }
        __syncthreads();
        if (active) {
            int koff = hi * 8;
            v16bf A2[3];
            #pragma unroll
            for (int kt=0; kt<3; ++kt) {
                const float* s = &sm[w][m][kt*32 + koff];
                #pragma unroll
                for (int j=0;j<8;++j) { A2[kt][j]=(bf16_t)s[j]; A2[kt][8+j]=(bf16_t)s[16+j]; }
            }
            #pragma unroll
            for (int nt=0; nt<NTN; ++nt) {
                v8f c = {};
                #pragma unroll
                for (int kt=0; kt<3; ++kt)
                    c = __builtin_amdgcn_wmma_f32_16x16x32_bf16(
                            false, A2[kt], false, lds_b_tile(sW2, kt, nt, lane),
                            (short)0, c, false, false);
                int n = nt*16 + m;
                float bb = b2[n];
                #pragma unroll
                for (int r=0;r<8;++r) {
                    float v = silu_f(c[r] + bb);
                    int rdst = srow[w][r + hi*8];
                    atomicAdd(agg + (size_t)rdst*FD + n, v);
                }
            }
        }
        __syncthreads();
    }
}

// --------------------------- fused node GCL ---------------------------------
// out = silu([h|agg] @ W1 + b1) @ W2 + b2   (persistent waves, LDS weights)
__global__ void __launch_bounds__(128)
k_node_gcl(const float* __restrict__ h, const float* __restrict__ agg,
           const bf16_t* __restrict__ W1, const float* __restrict__ b1,
           const bf16_t* __restrict__ W2, const float* __restrict__ b2,
           float* __restrict__ hout, int ntiles, int nwaves, int trips) {
    __shared__ bf16_t sW1[6*NTN*512];   // 36864 B
    __shared__ bf16_t sW2[3*NTN*512];   // 18432 B
    __shared__ float  sm[4][16][FD];
    if (threadIdx.x < 32) {
        tdm_load_1d(W1, sW1, (6*NTN*512*2) / 4);
        tdm_load_1d(W2, sW2, (3*NTN*512*2) / 4);
        __builtin_amdgcn_s_wait_tensorcnt(0);
    }
    __syncthreads();
    int w    = threadIdx.x >> 5;
    int lane = threadIdx.x & 31;
    int gw   = blockIdx.x * 4 + w;
    int m    = lane & 15;
    int hi   = lane >> 4;
    for (int t = 0; t < trips; ++t) {
        int tile = gw + t * nwaves;
        bool active = tile < ntiles;
        if (active) {
            int r0 = tile*16 + m;
            int koff = hi * 8;
            v16bf A[6];
            #pragma unroll
            for (int kt=0; kt<3; ++kt) {
                load_a_tile(h   + (size_t)r0*FD + kt*32 + koff, A[kt]);
                load_a_tile(agg + (size_t)r0*FD + kt*32 + koff, A[3+kt]);
            }
            #pragma unroll
            for (int nt=0; nt<NTN; ++nt) {
                v8f c = {};
                #pragma unroll
                for (int kt=0; kt<6; ++kt)
                    c = __builtin_amdgcn_wmma_f32_16x16x32_bf16(
                            false, A[kt], false, lds_b_tile(sW1, kt, nt, lane),
                            (short)0, c, false, false);
                int n = nt*16 + m;
                float bb = b1[n];
                #pragma unroll
                for (int r=0;r<8;++r) sm[w][r + hi*8][n] = silu_f(c[r] + bb);
            }
        }
        __syncthreads();
        if (active) {
            int koff = hi * 8;
            v16bf A2[3];
            #pragma unroll
            for (int kt=0; kt<3; ++kt) {
                const float* s = &sm[w][m][kt*32 + koff];
                #pragma unroll
                for (int j=0;j<8;++j) { A2[kt][j]=(bf16_t)s[j]; A2[kt][8+j]=(bf16_t)s[16+j]; }
            }
            #pragma unroll
            for (int nt=0; nt<NTN; ++nt) {
                v8f c = {};
                #pragma unroll
                for (int kt=0; kt<3; ++kt)
                    c = __builtin_amdgcn_wmma_f32_16x16x32_bf16(
                            false, A2[kt], false, lds_b_tile(sW2, kt, nt, lane),
                            (short)0, c, false, false);
                int n = nt*16 + m;
                float bb = b2[n];
                #pragma unroll
                for (int r=0;r<8;++r) {
                    int row = tile*16 + r + hi*8;
                    hout[(size_t)row*FD + n] = c[r] + bb;       // no activation
                }
            }
        }
        __syncthreads();
    }
}

// ------------------ decoder: silu(h@dw1+b1) @ dw2 + b2 ----------------------
__global__ void __launch_bounds__(128)
k_decoder(const float* __restrict__ h,
          const bf16_t* __restrict__ W1, const float* __restrict__ b1,
          const float* __restrict__ w2, const float* __restrict__ b2,
          float* __restrict__ dout, int ntiles, int nwaves, int trips) {
    __shared__ bf16_t sW1[3*NTN*512];
    __shared__ float  sm[4][16][FD];
    if (threadIdx.x < 32) {
        tdm_load_1d(W1, sW1, (3*NTN*512*2) / 4);
        __builtin_amdgcn_s_wait_tensorcnt(0);
    }
    __syncthreads();
    int w    = threadIdx.x >> 5;
    int lane = threadIdx.x & 31;
    int gw   = blockIdx.x * 4 + w;
    int m    = lane & 15;
    int hi   = lane >> 4;
    for (int t = 0; t < trips; ++t) {
        int tile = gw + t * nwaves;
        bool active = tile < ntiles;
        if (active) {
            int r0 = tile*16 + m;
            int koff = hi * 8;
            v16bf A[3];
            #pragma unroll
            for (int kt=0; kt<3; ++kt)
                load_a_tile(h + (size_t)r0*FD + kt*32 + koff, A[kt]);
            #pragma unroll
            for (int nt=0; nt<NTN; ++nt) {
                v8f c = {};
                #pragma unroll
                for (int kt=0; kt<3; ++kt)
                    c = __builtin_amdgcn_wmma_f32_16x16x32_bf16(
                            false, A[kt], false, lds_b_tile(sW1, kt, nt, lane),
                            (short)0, c, false, false);
                int n = nt*16 + m;
                float bb = b1[n];
                #pragma unroll
                for (int r=0;r<8;++r) sm[w][r + hi*8][n] = silu_f(c[r] + bb);
            }
        }
        __syncthreads();
        if (active && lane < 16) {
            int row = tile*16 + lane;
            #pragma unroll
            for (int j=0;j<3;++j) {
                float acc = b2[j];
                for (int k=0;k<FD;++k) acc += sm[w][lane][k] * w2[k*3 + j];
                dout[(size_t)row*3 + j] = acc;
            }
        }
        __syncthreads();
    }
}

// ---------- invert latent frame: mean over 8 frames, rotate back ------------
__global__ void k_invert_latent(const float* __restrict__ h, const float* __restrict__ Fb,
                                float* __restrict__ plat, int B) {
    int tid = blockIdx.x * blockDim.x + threadIdx.x;
    int total = B * NN * 32;
    if (tid >= total) return;
    int b   = tid / (NN*32);
    int rem = tid - b*(NN*32);
    int n   = rem / 32;
    int f   = rem - n*32;
    float t0=0.f, t1=0.f, t2=0.f;
    #pragma unroll
    for (int o=0;o<NO;++o) {
        const float* src = h + ((size_t)(o*B + b)*NN + n)*FD + f*3;
        t0 += sgn_oj(o,0) * src[0];
        t1 += sgn_oj(o,1) * src[1];
        t2 += sgn_oj(o,2) * src[2];
    }
    t0 *= 0.125f; t1 *= 0.125f; t2 *= 0.125f;
    const float* V = Fb + (size_t)b*9;
    float* dst = plat + ((size_t)b*(NN*32) + n*32 + f)*3;
    #pragma unroll
    for (int i=0;i<3;++i)
        dst[i] = V[i*3+0]*t0 + V[i*3+1]*t1 + V[i*3+2]*t2;
}

// -------- latent frame: covariance over 160 latent points, new eigh ---------
__global__ void k_latent_cov(const float* __restrict__ plat,
                             float* __restrict__ Fb, float* __restrict__ lctr, int B) {
    int b = blockIdx.x * blockDim.x + threadIdx.x;
    if (b >= B) return;
    const int M = NN*32;
    float s1[3] = {0.f,0.f,0.f};
    float s2[6] = {0.f,0.f,0.f,0.f,0.f,0.f};
    const float* p = plat + (size_t)b*M*3;
    for (int q=0;q<M;++q) {
        float x=p[q*3+0], y=p[q*3+1], z=p[q*3+2];
        s1[0]+=x; s1[1]+=y; s1[2]+=z;
        s2[0]+=x*x; s2[1]+=x*y; s2[2]+=x*z; s2[3]+=y*y; s2[4]+=y*z; s2[5]+=z*z;
    }
    float c0=s1[0]/M, c1=s1[1]/M, c2=s1[2]/M;
    float A[3][3], V[3][3];
    A[0][0]=s2[0]-M*c0*c0; A[0][1]=s2[1]-M*c0*c1; A[0][2]=s2[2]-M*c0*c2;
    A[1][0]=A[0][1];       A[1][1]=s2[3]-M*c1*c1; A[1][2]=s2[4]-M*c1*c2;
    A[2][0]=A[0][2];       A[2][1]=A[1][2];       A[2][2]=s2[5]-M*c2*c2;
    eigh3_asc(A, V);
    float* fb = Fb + (size_t)b*9;
    #pragma unroll
    for (int i=0;i<3;++i)
        #pragma unroll
        for (int j=0;j<3;++j) fb[i*3+j] = V[i][j];
    lctr[b*3+0]=c0; lctr[b*3+1]=c1; lctr[b*3+2]=c2;
}

// ---------- apply latent frames: fx = F^T (p - c) for all 8 frames ----------
__global__ void k_apply_latent(const float* __restrict__ plat,
                               const float* __restrict__ Fb, const float* __restrict__ lctr,
                               float* __restrict__ h, int B) {
    int tid = blockIdx.x * blockDim.x + threadIdx.x;
    int total = NO * B * NN;
    if (tid >= total) return;
    int o   = tid / (B*NN);
    int rem = tid - o*(B*NN);
    int b   = rem / NN;
    int n   = rem - b*NN;
    const float* V = Fb + (size_t)b*9;
    float c0=lctr[b*3+0], c1=lctr[b*3+1], c2=lctr[b*3+2];
    float s0=sgn_oj(o,0), s1=sgn_oj(o,1), s2=sgn_oj(o,2);
    float* dst = h + ((size_t)(o*B + b)*NN + n)*FD;
    for (int f=0; f<32; ++f) {
        const float* p = plat + ((size_t)b*(NN*32) + n*32 + f)*3;
        float q0=p[0]-c0, q1=p[1]-c1, q2=p[2]-c2;
        dst[f*3+0] = s0 * (V[0*3+0]*q0 + V[1*3+0]*q1 + V[2*3+0]*q2);
        dst[f*3+1] = s1 * (V[0*3+1]*q0 + V[1*3+1]*q1 + V[2*3+1]*q2);
        dst[f*3+2] = s2 * (V[0*3+2]*q0 + V[1*3+2]*q1 + V[2*3+2]*q2);
    }
}

// ------------- final invert frame: mean over frames + center ----------------
__global__ void k_invert_frame(const float* __restrict__ dec,
                               const float* __restrict__ Fb, const float* __restrict__ ctr,
                               float* __restrict__ out, int B) {
    int tid = blockIdx.x * blockDim.x + threadIdx.x;
    int total = B * NN;
    if (tid >= total) return;
    int b = tid / NN;
    int n = tid - b*NN;
    float t0=0.f, t1=0.f, t2=0.f;
    #pragma unroll
    for (int o=0;o<NO;++o) {
        const float* src = dec + ((size_t)(o*B + b)*NN + n)*3;
        t0 += sgn_oj(o,0) * src[0];
        t1 += sgn_oj(o,1) * src[1];
        t2 += sgn_oj(o,2) * src[2];
    }
    t0 *= 0.125f; t1 *= 0.125f; t2 *= 0.125f;
    const float* V = Fb + (size_t)b*9;
    #pragma unroll
    for (int i=0;i<3;++i)
        out[(size_t)tid*3 + i] = V[i*3+0]*t0 + V[i*3+1]*t1 + V[i*3+2]*t2 + ctr[b*3+i];
}

// ============================ host-side launch ==============================
static inline int cdiv_i(int a, int b) { return (a + b - 1) / b; }

extern "C" void kernel_launch(void* const* d_in, const int* in_sizes, int n_in,
                              void* d_out, int out_size, void* d_ws, size_t ws_size,
                              hipStream_t stream) {
    (void)n_in; (void)out_size; (void)ws_size;
    const float* hin  = (const float*)d_in[0];
    const float* ea   = (const float*)d_in[1];
    const float* embw = (const float*)d_in[2];
    const float* embb = (const float*)d_in[3];
    const float* ew1  = (const float*)d_in[4];
    const float* eb1  = (const float*)d_in[5];
    const float* ew2  = (const float*)d_in[6];
    const float* eb2  = (const float*)d_in[7];
    const float* nw1  = (const float*)d_in[8];
    const float* nb1  = (const float*)d_in[9];
    const float* nw2  = (const float*)d_in[10];
    const float* nb2  = (const float*)d_in[11];
    const float* dw1  = (const float*)d_in[12];
    const float* db1  = (const float*)d_in[13];
    const float* dw2  = (const float*)d_in[14];
    const float* db2  = (const float*)d_in[15];
    const int*   erow = (const int*)d_in[16];
    const int*   ecol = (const int*)d_in[17];

    const int NH   = in_sizes[0] / 6;          // 50000 nodes
    const int B    = NH / NN;                  // 10000 graphs
    const int E    = in_sizes[16];             // 200000 edges
    const int L    = in_sizes[4] / (194 * FD); // 4 layers
    const int ROWS = NO * NH;                  // 400000 frame rows
    const int ET   = (NO * E) / 16;            // edge tiles
    const int RT   = ROWS / 16;                // node tiles

    // persistent-wave sizing: nwaves multiple of 4 (waves per block)
    int nwaves = 12500;
    if (ET < nwaves) nwaves = ((ET + 3) / 4) * 4;
    const int eblk   = nwaves / 4;
    const int etrips = cdiv_i(ET, nwaves);
    int nwavesN = 12500;
    if (RT < nwavesN) nwavesN = ((RT + 3) / 4) * 4;
    const int nblk   = nwavesN / 4;
    const int ntrips = cdiv_i(RT, nwavesN);

    char* wp = (char*)d_ws;
    auto carve = [&](size_t bytes) -> char* {
        char* r = wp; wp += (bytes + 255) & ~(size_t)255; return r;
    };
    float*  hbuf = (float*)carve((size_t)ROWS * FD * 4);
    float*  h2   = (float*)carve((size_t)ROWS * FD * 4);
    float*  agg  = (float*)carve((size_t)ROWS * FD * 4);
    float*  Fb   = (float*)carve((size_t)B * 9 * 4);
    float*  ctr0 = (float*)carve((size_t)B * 3 * 4);
    float*  lctr = (float*)carve((size_t)B * 3 * 4);
    float*  plat = (float*)carve((size_t)B * (NN*32) * 3 * 4);
    float*  decb = (float*)carve((size_t)ROWS * 3 * 4);
    bf16_t* pew1 = (bf16_t*)carve((size_t)L * 7 * NTN * 512 * 2);
    bf16_t* pew2 = (bf16_t*)carve((size_t)L * 3 * NTN * 512 * 2);
    bf16_t* pnw1 = (bf16_t*)carve((size_t)L * 6 * NTN * 512 * 2);
    bf16_t* pnw2 = (bf16_t*)carve((size_t)L * 3 * NTN * 512 * 2);
    bf16_t* pdw1 = (bf16_t*)carve((size_t)3 * NTN * 512 * 2);

    // ---- pack all weights into WMMA B-operand layout (bf16) ----
    auto pack = [&](const float* src, bf16_t* dst, int K, int KT) {
        int total = KT * NTN * 512;
        k_pack_w<<<cdiv_i(total, 256), 256, 0, stream>>>(src, dst, K, KT);
    };
    for (int l = 0; l < L; ++l) {
        pack(ew1 + (size_t)l*194*FD, pew1 + (size_t)l*7*NTN*512, 194, 7);
        pack(ew2 + (size_t)l*FD*FD,  pew2 + (size_t)l*3*NTN*512,  96, 3);
        pack(nw1 + (size_t)l*192*FD, pnw1 + (size_t)l*6*NTN*512, 192, 6);
        pack(nw2 + (size_t)l*FD*FD,  pnw2 + (size_t)l*3*NTN*512,  96, 3);
    }
    pack(dw1, pdw1, 96, 3);

    // ---- initial frames + embedding ----
    k_make_frame<<<cdiv_i(B, 128), 128, 0, stream>>>(hin, Fb, ctr0, B);
    k_frame_embed<<<cdiv_i(ROWS, 128), 128, 0, stream>>>(hin, Fb, ctr0, embw, embb,
                                                         hbuf, B, ROWS);

    // ---- GCL layers ----
    for (int l = 0; l < L; ++l) {
        k_zero<<<cdiv_i(ROWS * FD, 256), 256, 0, stream>>>(agg, ROWS * FD);
        k_edge_gcl<<<eblk, 128, 0, stream>>>(
            hbuf, erow, ecol, ea,
            pew1 + (size_t)l*7*NTN*512, eb1 + (size_t)l*FD,
            pew2 + (size_t)l*3*NTN*512, eb2 + (size_t)l*FD,
            agg, E, NH, ET, nwaves, etrips);
        k_node_gcl<<<nblk, 128, 0, stream>>>(
            hbuf, agg,
            pnw1 + (size_t)l*6*NTN*512, nb1 + (size_t)l*FD,
            pnw2 + (size_t)l*3*NTN*512, nb2 + (size_t)l*FD,
            h2, RT, nwavesN, ntrips);
        if (l < L - 1) {
            k_invert_latent<<<cdiv_i(B*NN*32, 256), 256, 0, stream>>>(h2, Fb, plat, B);
            k_latent_cov<<<cdiv_i(B, 128), 128, 0, stream>>>(plat, Fb, lctr, B);
            k_apply_latent<<<cdiv_i(NO*B*NN, 256), 256, 0, stream>>>(plat, Fb, lctr,
                                                                     hbuf, B);
        }
    }

    // ---- decoder + final frame inversion ----
    k_decoder<<<nblk, 128, 0, stream>>>(h2, pdw1, db1, dw2, db2, decb,
                                        RT, nwavesN, ntrips);
    k_invert_frame<<<cdiv_i(B*NN, 256), 256, 0, stream>>>(decb, Fb, ctr0,
                                                          (float*)d_out, B);
}
